// ModalityMoE_63264868270614
// MI455X (gfx1250) — compile-verified, matrix operand order checked by reading, not run on hardware
//
#include <hip/hip_runtime.h>
#include <hip/hip_bf16.h>
#include <stdint.h>

typedef __attribute__((ext_vector_type(16))) __bf16 v16bf;
typedef __attribute__((ext_vector_type(8)))  float  v8f;
typedef int v4i_ __attribute__((__vector_size__(16)));
typedef __attribute__((address_space(1))) v4i_* gp_v4i;
typedef __attribute__((address_space(3))) v4i_* lp_v4i;

#define B_ 8
#define L_ 3072
#define D_ 256
#define E_ 4
#define H_ 1024

// d_ws layout (bytes)
#define AGG_OFF   0                     // B*3D floats = 24 KiB
#define WBE_OFF   (32*1024)             // B*E floats
#define PACK_EW1  (64*1024)             // 4 experts * 512 tiles * 1 KiB = 2 MiB
#define PACK_EW2  (PACK_EW1 + 2*1024*1024)
#define PACK_SW1  (PACK_EW2 + 2*1024*1024)   // 512 KiB
#define PACK_SW2  (PACK_SW1 + 512*1024)      // 512 KiB

#if defined(__has_builtin)
# if __has_builtin(__builtin_amdgcn_global_load_async_to_lds_b128)
#  define HAVE_ASYNC 1
# endif
#endif

__device__ __forceinline__ uint16_t f2bfbits(float f) {
    union { float f; uint32_t u; } a; a.f = f;
    uint32_t r = a.u + 0x7FFFu + ((a.u >> 16) & 1u);   // round-to-nearest-even
    return (uint16_t)(r >> 16);
}
__device__ __forceinline__ __bf16 f2bf(float f) {
    uint16_t h = f2bfbits(f);
    __bf16 o; __builtin_memcpy(&o, &h, 2); return o;
}
__device__ __forceinline__ float gelu_t(float x) {   // tanh-approx gelu (approximate=True)
    return 0.5f * x * (1.0f + tanhf(0.7978845608028654f * (x + 0.044715f * x * x * x)));
}

// 16-byte global -> LDS copy: async (ASYNCcnt, bypasses VGPRs) when available.
__device__ __forceinline__ void cp16(const char* g, char* l) {
#if HAVE_ASYNC
    __builtin_amdgcn_global_load_async_to_lds_b128(
        (gp_v4i)(const_cast<char*>(g)), (lp_v4i)(l), 0, 0);
#else
    *(float4*)l = *(const float4*)g;
#endif
}
__device__ __forceinline__ void cp_wait() {
#if HAVE_ASYNC
# if __has_builtin(__builtin_amdgcn_s_wait_asynccnt)
    __builtin_amdgcn_s_wait_asynccnt(0);
# else
    asm volatile("s_wait_asynccnt 0" ::: "memory");
# endif
#endif
}

// ---------------------------------------------------------------------------
// Kernel 1: segment aggregates -> gate_in (full / head+prop / wrist+prop)
// ---------------------------------------------------------------------------
__global__ __launch_bounds__(256) void agg_kernel(const float* __restrict__ x,
                                                  float* __restrict__ agg) {
    int b = blockIdx.x, d = threadIdx.x;
    const float* xb = x + (size_t)b * L_ * D_ + d;
    float hs = 0.f, wsu = 0.f, ps = 0.f;
    for (int l = 0;    l < 1024; ++l) hs  += xb[(size_t)l * D_];
    for (int l = 1024; l < 2048; ++l) wsu += xb[(size_t)l * D_];
    for (int l = 2048; l < 3072; ++l) ps  += xb[(size_t)l * D_];
    agg[b * 768 + d]       = (hs + wsu + ps) * (1.0f / 3072.0f);
    agg[b * 768 + 256 + d] = (hs + ps)       * (1.0f / 2048.0f);
    agg[b * 768 + 512 + d] = (wsu + ps)      * (1.0f / 2048.0f);
}

// ---------------------------------------------------------------------------
// Kernel 2: gating -> w_be (renormalized top-2, zeros elsewhere)
// ---------------------------------------------------------------------------
__global__ __launch_bounds__(64) void gate_kernel(const float* __restrict__ agg,
                                                  const float* __restrict__ tc,
                                                  const float* __restrict__ gw,
                                                  const float* __restrict__ gb,
                                                  const float* __restrict__ tw,
                                                  const float* __restrict__ tb,
                                                  float* __restrict__ wbe) {
    __shared__ float lg[B_][E_];
    __shared__ float md[B_][2 * E_];
    int t = threadIdx.x;
    if (t < B_ * E_) {
        int b = t >> 2, e = t & 3;
        float s = gb[e];
        for (int k = 0; k < 3 * D_; ++k) s += agg[b * 3 * D_ + k] * gw[k * E_ + e];
        lg[b][e] = s;
    }
    {   // 64 threads: (b, j) modulation
        int b = t >> 3, j = t & 7;
        float s = tb[j];
        for (int d = 0; d < D_; ++d) {
            float v = tc[b * D_ + d];
            s += (v / (1.0f + expf(-v))) * tw[d * 2 * E_ + j];   // silu
        }
        md[b][j] = s;
    }
    __syncthreads();
    if (t < B_) {
        int b = t;
        float p[E_]; float mx = -1e30f;
        for (int e = 0; e < E_; ++e) {
            p[e] = lg[b][e] * (1.0f + md[b][e]) + md[b][e + E_];
            mx = fmaxf(mx, p[e]);
        }
        float se = 0.f;
        for (int e = 0; e < E_; ++e) { p[e] = expf(p[e] - mx); se += p[e]; }
        for (int e = 0; e < E_; ++e) p[e] /= se;
        int i1 = 0;
        for (int e = 1; e < E_; ++e) if (p[e] > p[i1]) i1 = e;
        int i2 = (i1 == 0) ? 1 : 0;
        for (int e = 0; e < E_; ++e) if (e != i1 && p[e] > p[i2]) i2 = e;
        float s2 = p[i1] + p[i2] + 1e-8f;
        for (int e = 0; e < E_; ++e)
            wbe[b * E_ + e] = (e == i1) ? (p[i1] / s2) : ((e == i2) ? (p[i2] / s2) : 0.0f);
    }
}

// ---------------------------------------------------------------------------
// Kernel 3: pack fp32 weights into bf16 WMMA B-operand tiles.
// Tile = 32(K) x 16(N); per tile, lane l owns 16 bf16 at byte offset l*32.
// Element e of lane l maps to (kk, n):
//   n = l & 15 ; kk = (e/8)*16 + (l/16)*8 + (e%8)
// Tiles stored [nt][kt] contiguous in kt.
// ---------------------------------------------------------------------------
__global__ __launch_bounds__(128) void pack_kernel(const float* __restrict__ ew1,
                                                   const float* __restrict__ ew2,
                                                   const float* __restrict__ sw1,
                                                   const float* __restrict__ sw2,
                                                   char* __restrict__ ws) {
    __shared__ float tileS[512];
    int tb = blockIdx.x;
    const float* src; char* dst; int K, N, local;
    if (tb < 2048) {           // ew1: (E, 256, 1024), K=256 N=1024, 512 tiles/expert
        int ex = tb >> 9; local = tb & 511;
        src = ew1 + (size_t)ex * D_ * H_;
        dst = ws + PACK_EW1 + (size_t)tb * 1024;
        K = D_; N = H_;
    } else if (tb < 4096) {    // ew2: (E, 1024, 256), K=1024 N=256
        int t2 = tb - 2048; int ex = t2 >> 9; local = t2 & 511;
        src = ew2 + (size_t)ex * H_ * D_;
        dst = ws + PACK_EW2 + (size_t)t2 * 1024;
        K = H_; N = D_;
    } else if (tb < 4608) {    // sw1: (256, 1024)
        local = tb - 4096;
        src = sw1; dst = ws + PACK_SW1 + (size_t)local * 1024;
        K = D_; N = H_;
    } else {                   // sw2: (1024, 256)
        local = tb - 4608;
        src = sw2; dst = ws + PACK_SW2 + (size_t)local * 1024;
        K = H_; N = D_;
    }
    int ktiles = K / 32;
    int nt = local / ktiles, kt = local % ktiles;
    int k0 = kt * 32, n0 = nt * 16;
    int t = threadIdx.x;
    #pragma unroll
    for (int i = 0; i < 4; ++i) {
        int e = t + 128 * i;
        int k = e >> 4, nn = e & 15;
        tileS[e] = src[(size_t)(k0 + k) * N + (n0 + nn)];
    }
    __syncthreads();
    uint16_t hv[4];
    #pragma unroll
    for (int i = 0; i < 4; ++i) {
        int p = t * 4 + i;
        int lanep = p >> 4, el = p & 15;
        int nn = lanep & 15, half = lanep >> 4;
        int kk = ((el >> 3) << 4) + (half << 3) + (el & 7);
        hv[i] = f2bfbits(tileS[kk * 16 + nn]);
    }
    uint32_t lo = (uint32_t)hv[0] | ((uint32_t)hv[1] << 16);
    uint32_t hi = (uint32_t)hv[2] | ((uint32_t)hv[3] << 16);
    ((uint2*)dst)[t] = make_uint2(lo, hi);
}

// ---------------------------------------------------------------------------
// Kernel 4: fused MoE + shared MLP.
// Block = 256 thr = 8 waves; each wave owns a 16-row token tile (block: 128 rows,
// all inside one segment -> expert activity is block-uniform).
// Per n2c iteration the whole block shares one staged copy of the 32 B-tiles
// (16 GEMM1 + 16 GEMM2 = 32 KiB), double-buffered via async global->LDS copies.
// Inner GEMM loops software-pipeline the B-tile ds_loads one tile ahead so the
// DS latency overlaps the XDL (WMMA) pipe instead of serializing on dscnt==0.
// ---------------------------------------------------------------------------
__global__ __launch_bounds__(256) void moe_kernel(const float* __restrict__ x,
                                                  const float* __restrict__ eb1,
                                                  const float* __restrict__ eb2,
                                                  const float* __restrict__ sb1,
                                                  const float* __restrict__ sb2,
                                                  const float* __restrict__ wbe,
                                                  const char* __restrict__ ws,
                                                  float* __restrict__ out) {
    __shared__ float cbias[D_];
    __shared__ float wz[8];
    __shared__ float b1s[H_];
    __shared__ __align__(32) uint16_t hstage[8][16 * 32];
    __shared__ __align__(32) char bstage[2][32 * 1024];   // [buf][tile 0..31][1KiB]

    const int tid   = threadIdx.x;
    const int wv    = tid >> 5;
    const int lane  = tid & 31;
    const int n     = lane & 15;     // B/C column ; A row
    const int khalf = lane >> 4;
    const int b     = blockIdx.y;
    const int l0    = blockIdx.x * 128 + wv * 16;
    const int seg   = (blockIdx.x * 128) >> 10;  // 0=head 1=wrist 2=prop

    if (tid < E_) {
        float w = wbe[b * E_ + tid];
        bool act = (tid == 1) ? (seg != 1) : (tid == 2) ? (seg != 0) : true;
        wz[tid] = act ? w : 0.0f;
    }
    if (tid == E_) wz[E_] = 1.0f;    // shared MLP pseudo-expert
    __syncthreads();
    {   // combined output bias: sb2 + sum_e w_e * eb2_e  (blockDim == 256 == D_)
        float s = sb2[tid];
        #pragma unroll
        for (int e = 0; e < E_; ++e) s += wz[e] * eb2[e * D_ + tid];
        cbias[tid] = s;
    }

    // Build A1: x rows [l0, l0+16) -> bf16 A-operand layout, 8 K-chunks of 32
    const float4* xr = (const float4*)(x + ((size_t)b * L_ + l0 + n) * D_);
    v16bf A1[8];
    #pragma unroll
    for (int c = 0; c < 8; ++c) {
        #pragma unroll
        for (int part = 0; part < 2; ++part) {
            int kb = c * 32 + part * 16 + khalf * 8;
            float4 f0 = xr[kb >> 2];
            float4 f1 = xr[(kb >> 2) + 1];
            A1[c][part * 8 + 0] = f2bf(f0.x);
            A1[c][part * 8 + 1] = f2bf(f0.y);
            A1[c][part * 8 + 2] = f2bf(f0.z);
            A1[c][part * 8 + 3] = f2bf(f0.w);
            A1[c][part * 8 + 4] = f2bf(f1.x);
            A1[c][part * 8 + 5] = f2bf(f1.y);
            A1[c][part * 8 + 6] = f2bf(f1.z);
            A1[c][part * 8 + 7] = f2bf(f1.w);
        }
    }

    v8f outAcc[16];
    #pragma unroll
    for (int i = 0; i < 16; ++i) { v8f z = {}; outAcc[i] = z; }

    for (int ex = 0; ex < 5; ++ex) {
        float wgt = wz[ex];
        if (wgt == 0.0f) continue;   // block-uniform: skip masked / non-top-k experts
        const char* w1p = (ex < 4) ? (ws + PACK_EW1 + (size_t)ex * 512 * 1024)
                                   : (ws + PACK_SW1);
        const char* w2p = (ex < 4) ? (ws + PACK_EW2 + (size_t)ex * 512 * 1024)
                                   : (ws + PACK_SW2);
        const float* b1p = (ex < 4) ? (eb1 + ex * H_) : sb1;

        __syncthreads();   // prev expert fully done before we touch b1s / bstage[0]
        for (int i = tid; i < H_; i += 256) b1s[i] = b1p[i];

        // cooperative stage of the 32 B tiles for one n2c: 128 bytes per thread
        auto stage = [&](int n2c, int bb) {
            char* dst = bstage[bb];
            if (tid < 128) {    // GEMM1 tiles: 16 KiB contiguous at w1p + n2c*16K
                const char* srcp = w1p + (size_t)n2c * 16384 + (size_t)tid * 128;
                char* d = dst + (size_t)tid * 128;
                #pragma unroll
                for (int i = 0; i < 8; ++i) cp16(srcp + i * 16, d + i * 16);
            } else {            // GEMM2 tiles: n3 = u/8, 128B chunk u%8
                int u = tid - 128;
                int n3 = u >> 3, ch = u & 7;
                const char* srcp = w2p + ((size_t)(n3 * 32 + n2c)) * 1024 + (size_t)ch * 128;
                char* d = dst + 16384 + (size_t)u * 128;
                #pragma unroll
                for (int i = 0; i < 8; ++i) cp16(srcp + i * 16, d + i * 16);
            }
        };

        stage(0, 0);
        for (int n2c = 0; n2c < 32; ++n2c) {       // H in chunks of 32
            const char* buf = bstage[n2c & 1];
            cp_wait();                 // own async copies landed
            __syncthreads();           // everyone's copies landed; prev compute done
            if (n2c + 1 < 32) stage(n2c + 1, (n2c + 1) & 1);

            // ---- GEMM1: h columns [n2c*32, n2c*32+32) ----
            #pragma unroll
            for (int t = 0; t < 2; ++t) {
                v8f acc = {};
                v16bf btc, btn;
                __builtin_memcpy(&btc, buf + (size_t)(t * 8) * 1024 + lane * 32, 32);
                #pragma unroll
                for (int kc = 0; kc < 8; ++kc) {   // K = D = 256, pipelined B loads
                    if (kc + 1 < 8)
                        __builtin_memcpy(&btn, buf + (size_t)(t * 8 + kc + 1) * 1024 + lane * 32, 32);
                    acc = __builtin_amdgcn_wmma_f32_16x16x32_bf16(
                              false, A1[kc], false, btc, (short)0, acc, false, false);
                    btc = btn;
                }
                float bias = b1s[(n2c * 2 + t) * 16 + n];
                #pragma unroll
                for (int r = 0; r < 8; ++r) {      // bias + gelu + weight, restage (C->A)
                    float hv = gelu_t(acc[r] + bias) * wgt;
                    int mrow = r + 8 * khalf;
                    int kk   = t * 16 + n;
                    int lp   = mrow + 16 * ((kk >> 3) & 1);
                    int e2   = ((kk >> 4) << 3) + (kk & 7);
                    hstage[wv][lp * 16 + e2] = f2bfbits(hv);
                }
            }
            v16bf a2;
            __builtin_memcpy(&a2, &hstage[wv][lane * 16], 32);  // per-wave slice, DS in-order
            // ---- GEMM2: outAcc += a2(16x32) @ W2[n2c*32 .. , :256), pipelined ----
            {
                v16bf btc, btn;
                __builtin_memcpy(&btc, buf + 16384 + (size_t)lane * 32, 32);
                #pragma unroll
                for (int n3 = 0; n3 < 16; ++n3) {
                    if (n3 + 1 < 16)
                        __builtin_memcpy(&btn, buf + 16384 + (size_t)(n3 + 1) * 1024 + lane * 32, 32);
                    outAcc[n3] = __builtin_amdgcn_wmma_f32_16x16x32_bf16(
                                     false, a2, false, btc, (short)0, outAcc[n3], false, false);
                    btc = btn;
                }
            }
        }
    }
    __syncthreads();

    // Write output tile (+ combined bias)
    #pragma unroll
    for (int n3 = 0; n3 < 16; ++n3) {
        #pragma unroll
        for (int r = 0; r < 8; ++r) {
            int row = r + 8 * khalf;
            int col = n3 * 16 + n;
            out[((size_t)b * L_ + l0 + row) * D_ + col] = outAcc[n3][r] + cbias[col];
        }
    }
}

// ---------------------------------------------------------------------------
extern "C" void kernel_launch(void* const* d_in, const int* in_sizes, int n_in,
                              void* d_out, int out_size, void* d_ws, size_t ws_size,
                              hipStream_t stream) {
    (void)in_sizes; (void)n_in; (void)out_size; (void)ws_size;
    const float* x   = (const float*)d_in[0];
    const float* tc  = (const float*)d_in[1];
    const float* gw  = (const float*)d_in[2];
    const float* gb  = (const float*)d_in[3];
    const float* tw  = (const float*)d_in[4];
    const float* tb  = (const float*)d_in[5];
    const float* ew1 = (const float*)d_in[6];
    const float* eb1 = (const float*)d_in[7];
    const float* ew2 = (const float*)d_in[8];
    const float* eb2 = (const float*)d_in[9];
    const float* sw1 = (const float*)d_in[10];
    const float* sb1 = (const float*)d_in[11];
    const float* sw2 = (const float*)d_in[12];
    const float* sb2 = (const float*)d_in[13];
    char*  ws  = (char*)d_ws;
    float* agg = (float*)(ws + AGG_OFF);
    float* wbe = (float*)(ws + WBE_OFF);
    float* out = (float*)d_out;

    agg_kernel <<<B_, 256, 0, stream>>>(x, agg);
    gate_kernel<<<1, 64, 0, stream>>>(agg, tc, gw, gb, tw, tb, wbe);
    pack_kernel<<<5120, 128, 0, stream>>>(ew1, ew2, sw1, sw2, ws);
    moe_kernel <<<dim3(L_ / 128, B_), 256, 0, stream>>>(x, eb1, eb2, sb1, sb2, wbe, ws, out);
}
